// VectorQuantiser_9474697855751
// MI455X (gfx1250) — matrix operand order repaired and checked by reading, compiled, not run on hardware
//
#include <hip/hip_runtime.h>

// ---------------- problem constants ----------------
#define B_     16
#define CIN    256
#define HW     4096          // 64*64
#define D_     128
#define K_     1024
#define N_     (B_ * HW)     // 65536 pixels
#define TILE_M 128           // pixels per workgroup
#define NWG    (N_ / TILE_M) // 512

#define XPAD   264           // bf16 elems per LDS x-row  (528 B, 16B aligned)
#define ZPAD   132           // f32  elems per LDS z-row  (528 B, same stride -> clean alias)

#define ECHUNK_ROWS 128      // codes per staged Et chunk
#define EROWB  272           // LDS row stride for staged Et: 256 B data + 16 B TDM pad
#define NCHUNK (K_ / ECHUNK_ROWS)  // 8

typedef __attribute__((ext_vector_type(16))) __bf16 v16bf;
typedef __attribute__((ext_vector_type(8)))  float  v8f;

union Frag {               // one WMMA 16-bit A/B operand: 32 B per lane
    v16bf v;
    uint4 q[2];
};

// ---------------- TDM availability ----------------
#if defined(__has_builtin)
#if __has_builtin(__builtin_amdgcn_tensor_load_to_lds) && \
    __has_builtin(__builtin_amdgcn_s_wait_tensorcnt)
#define HAVE_TDM 1
#endif
#endif
#ifndef HAVE_TDM
#define HAVE_TDM 0
#endif

#if HAVE_TDM
typedef unsigned int tdm_v4u __attribute__((ext_vector_type(4)));
typedef int          tdm_v4i __attribute__((ext_vector_type(4)));
typedef int          tdm_v8i __attribute__((ext_vector_type(8)));

// DMA one contiguous 32 KB Et chunk (128 rows x 256 B) into LDS, inserting
// 16 B of padding after every 256 B row (pad_interval=code 5 -> 32x8B=256 B,
// pad_amount=code 3 -> 4 dwords = 16 B)  => 272 B LDS row stride.
__device__ __forceinline__ void tdm_issue_chunk(const __bf16* Et, int ch, void* ldsdst)
{
    unsigned long long ga  = (unsigned long long)(uintptr_t)(Et + (size_t)ch * ECHUNK_ROWS * D_);
    unsigned int       lds = (unsigned int)(uintptr_t)ldsdst;   // low 32 bits = LDS offset

    tdm_v4u g0 = { 1u,                                   // count=1, is_restore=0
                   lds,                                  // lds_addr
                   (unsigned int)ga,                     // global_addr[31:0]
                   (unsigned int)(ga >> 32) | (2u << 30) /* type=2 */ };

    tdm_v8i g1 = { (int)((3u << 16)      // data_size = 3 (8 B)
                       | (1u << 20)      // pad_enable
                       | (5u << 22)      // pad_interval code 5 (256 B)
                       | (3u << 25)),    // pad_amount  code 3 (16 B)
                   (int)(32u  << 16),    // tensor_dim0 = 32 (8B units / row)
                   (int)(128u << 16),    // tensor_dim1 = 128 rows
                   (int)(32u  << 16),    // tile_dim0 = 32
                   128,                  // tile_dim1 = 128, tile_dim2 = 0
                   32,                   // tensor_dim0_stride = 32 (contiguous rows)
                   0, 0 };

    tdm_v4i gz = { 0, 0, 0, 0 };
#if __clang_major__ >= 23
    tdm_v8i gz8 = { 0, 0, 0, 0, 0, 0, 0, 0 };
    __builtin_amdgcn_tensor_load_to_lds(g0, g1, gz, gz, gz8, 0);
#else
    __builtin_amdgcn_tensor_load_to_lds(g0, g1, gz, gz, 0);
#endif
}
#endif // HAVE_TDM

// ---------------- prep: transpose + bf16 convert ----------------
// Wt[d][c] = bf16(conv_w[c][d])   (64 KB)
// Et[k][d] = bf16(embed[d][k])    (256 KB)
__global__ __launch_bounds__(256) void vq_prep(const float* __restrict__ conv_w,
                                               const float* __restrict__ embed,
                                               __bf16* __restrict__ Wt,
                                               __bf16* __restrict__ Et)
{
    int i = blockIdx.x * 256 + threadIdx.x;
    if (i < CIN * D_) {
        int c = i / D_, d = i % D_;
        Wt[(size_t)d * CIN + c] = (__bf16)conv_w[i];
    }
    if (i < D_ * K_) {
        int d = i / K_, k = i % K_;
        Et[(size_t)k * D_ + d] = (__bf16)embed[i];
    }
}

// Enorm[k] = sum_d embed[d][k]^2   (reads coalesced across k)
__global__ __launch_bounds__(256) void vq_enorm(const float* __restrict__ embed,
                                                float* __restrict__ Enorm)
{
    int k = blockIdx.x * 256 + threadIdx.x;   // 1024 total
    float s = 0.f;
    for (int d = 0; d < D_; ++d) {
        float e = embed[(size_t)d * K_ + k];
        s = fmaf(e, e, s);
    }
    Enorm[k] = s;
}

// ---------------- fused main kernel ----------------
__global__ __launch_bounds__(256) void vq_main(const float* __restrict__ x,
                                               const float* __restrict__ conv_b,
                                               const float* __restrict__ embed,
                                               const __bf16* __restrict__ Wt,
                                               const __bf16* __restrict__ Et,
                                               const float* __restrict__ Enorm,
                                               float* __restrict__ q_out,
                                               float* __restrict__ ind_out,
                                               float* __restrict__ partials)
{
    // x tile (bf16, [m][c], XPAD) aliases z tile (f32, [m][d], ZPAD): both 528 B/row
    __shared__ __align__(16) unsigned char lds_buf[TILE_M * XPAD * 2];      // 67584 B
    __shared__ __align__(16) unsigned char lds_e[2][ECHUNK_ROWS * EROWB];   // 2 x 34816 B
    __shared__ int   lds_idx[TILE_M];
    __shared__ float lds_red[256];

    __bf16* lds_x = (__bf16*)lds_buf;
    float*  lds_z = (float*)lds_buf;

    const int t    = threadIdx.x;
    const int wg   = blockIdx.x;
    const int n0   = wg * TILE_M;        // first pixel of tile
    const int b    = n0 >> 12;           // /HW
    const int hw0  = n0 & (HW - 1);
    const int lane = t & 31;
    const int wv   = t >> 5;             // wave 0..7
    const int col  = lane & 15;          // N-column within 16x16 tile
    const int hi   = lane >> 4;          // lane half
    const int koff = hi * 8;             // A-fragment K sub-offset
    const int mrow = wv * 16 + col;      // A-fragment row (pixel) for this lane

    const v8f zero = {0.f, 0.f, 0.f, 0.f, 0.f, 0.f, 0.f, 0.f};

    // ---- Phase 1: x (NCHW fp32) -> LDS bf16 [m][c] ----
    {
        const float* xb = x + (size_t)b * CIN * HW + hw0;
        for (int i = t; i < TILE_M * CIN / 4; i += 256) {
            int c  = i >> 5;             // 32 float4 per channel row
            int m4 = (i & 31) << 2;
            const float4 v = *(const float4*)(xb + (size_t)c * HW + m4);
            __bf16* dst = lds_x + (size_t)m4 * XPAD + c;
            dst[0 * XPAD] = (__bf16)v.x;
            dst[1 * XPAD] = (__bf16)v.y;
            dst[2 * XPAD] = (__bf16)v.z;
            dst[3 * XPAD] = (__bf16)v.w;
        }
    }
    __syncthreads();

    // ---- Phase 2: GEMM1  z[128][128] = x[128][256] * Wt^T ----
    // Flattened kt x n loop with 1-deep register prefetch of next A/B fragments.
    v8f acc[8];
#pragma unroll
    for (int n = 0; n < 8; ++n) acc[n] = zero;
    {
        const __bf16* abase = lds_x + (size_t)mrow * XPAD + koff;
        const __bf16* wbase = Wt + (size_t)col * CIN + hi * 8;

        Frag a_cur, b_cur;
        a_cur.q[0] = *(const uint4*)(abase);
        a_cur.q[1] = *(const uint4*)(abase + 16);
        b_cur.q[0] = *(const uint4*)(wbase);
        b_cur.q[1] = *(const uint4*)(wbase + 16);

#pragma unroll
        for (int it = 0; it < 64; ++it) {
            const int n = it & 7;
            Frag a_nxt, b_nxt;
            if (it < 63) {
                const int it2 = it + 1;
                const int kt2 = it2 >> 3, n2 = it2 & 7;
                const __bf16* bp = wbase + (size_t)(n2 * 16) * CIN + kt2 * 32;
                b_nxt.q[0] = *(const uint4*)(bp);
                b_nxt.q[1] = *(const uint4*)(bp + 16);
                if (n2 == 0) {
                    const __bf16* ap = abase + kt2 * 32;
                    a_nxt.q[0] = *(const uint4*)(ap);
                    a_nxt.q[1] = *(const uint4*)(ap + 16);
                }
            }
            acc[n] = __builtin_amdgcn_wmma_f32_16x16x32_bf16(
                false, a_cur.v, false, b_cur.v, (short)0, acc[n], false, false);
            if (it < 63) {
                b_cur = b_nxt;
                if (((it + 1) & 7) == 0) a_cur = a_nxt;
            }
        }
    }
    __syncthreads();   // all waves done reading x before z overwrites the region

    // ---- Phase 2b: bias + store z (fp32) into LDS ----
    {
        const int m0 = wv * 16 + hi * 8;          // C/D row base for this lane half
#pragma unroll
        for (int n = 0; n < 8; ++n) {
            float bn = conv_b[n * 16 + col];
#pragma unroll
            for (int r = 0; r < 8; ++r)
                lds_z[(size_t)(m0 + r) * ZPAD + n * 16 + col] = acc[n][r] + bn;
        }
    }
    __syncthreads();

    // ---- Phase 3: GEMM2 + fused argmin over K=1024 codes ----
    // score = -2*(z . E_k) + ||E_k||^2   (||z||^2 constant per row: dropped)
    Frag za[4];
    {
        const float* zr = lds_z + (size_t)mrow * ZPAD;
#pragma unroll
        for (int kt = 0; kt < 4; ++kt) {
            const float* p = zr + kt * 32 + koff;
#pragma unroll
            for (int e = 0; e < 8; ++e) {
                za[kt].v[e]     = (__bf16)p[e];
                za[kt].v[e + 8] = (__bf16)p[e + 16];
            }
        }
    }
    float minv[8];
    int   mini[8];
#pragma unroll
    for (int r = 0; r < 8; ++r) { minv[r] = 3.4e38f; mini[r] = 0; }

#if HAVE_TDM
    if (wv == 0) tdm_issue_chunk(Et, 0, &lds_e[0][0]);   // prologue: chunk 0 in flight
#endif

    for (int ch = 0; ch < NCHUNK; ++ch) {
#if HAVE_TDM
        if (wv == 0) {
            if (ch + 1 < NCHUNK) {
                tdm_issue_chunk(Et, ch + 1, &lds_e[(ch + 1) & 1][0]);
                __builtin_amdgcn_s_wait_tensorcnt(1);   // chunk ch landed, ch+1 in flight
            } else {
                __builtin_amdgcn_s_wait_tensorcnt(0);   // last chunk landed
            }
        }
        __syncthreads();                         // publish chunk ch to all waves
        const unsigned char* eb = &lds_e[ch & 1][0];
#else
        __syncthreads();                         // prior consumers done with buffer
        {   // cooperative fallback copy: global b128 -> ds b128, 272 B row stride
            const unsigned char* src = (const unsigned char*)(Et + (size_t)ch * ECHUNK_ROWS * D_);
            for (int i = t; i < ECHUNK_ROWS * 16; i += 256) {
                int row = i >> 4, jj = i & 15;
                *(uint4*)(&lds_e[0][0] + (size_t)row * EROWB + jj * 16) =
                    *(const uint4*)(src + (size_t)row * 256 + jj * 16);
            }
        }
        __syncthreads();
        const unsigned char* eb = &lds_e[0][0];
#endif
        for (int nt = 0; nt < 8; ++nt) {
            Frag bf4[4];
            const unsigned char* er = eb + (size_t)(nt * 16 + col) * EROWB + hi * 16;
#pragma unroll
            for (int kt = 0; kt < 4; ++kt) {     // 8 ds_load_b128, one wait
                bf4[kt].q[0] = *(const uint4*)(er + kt * 64);
                bf4[kt].q[1] = *(const uint4*)(er + kt * 64 + 32);
            }
            v8f c = zero;
#pragma unroll
            for (int kt = 0; kt < 4; ++kt)
                c = __builtin_amdgcn_wmma_f32_16x16x32_bf16(
                    false, za[kt].v, false, bf4[kt].v, (short)0, c, false, false);

            const int   ncode = ch * ECHUNK_ROWS + nt * 16 + col;
            const float en    = Enorm[ncode];
#pragma unroll
            for (int r = 0; r < 8; ++r) {
                float s = fmaf(-2.f, c[r], en);
                if (s < minv[r]) { minv[r] = s; mini[r] = ncode; }
            }
        }
        __syncthreads();                         // buffer ch may be overwritten next iter
    }

    // butterfly argmin across the 16 lanes holding one pixel's columns
#pragma unroll
    for (int off = 1; off < 16; off <<= 1) {
#pragma unroll
        for (int r = 0; r < 8; ++r) {
            float ov = __shfl_xor(minv[r], off, 32);
            int   oi = __shfl_xor(mini[r], off, 32);
            if (ov < minv[r] || (ov == minv[r] && oi < mini[r])) {
                minv[r] = ov; mini[r] = oi;
            }
        }
    }
    if (col == 0) {
        const int m0 = wv * 16 + hi * 8;
#pragma unroll
        for (int r = 0; r < 8; ++r) {
            lds_idx[m0 + r] = mini[r];
            ind_out[n0 + m0 + r] = (float)mini[r];
        }
    }
    __syncthreads();

    // ---- Phase 4: gather exact fp32 codebook rows, write NCHW, diff partial ----
    float dsum = 0.f;
    {
        const int m  = t & 127;            // pixel within tile
        const int dg = (t >> 7) * 64;      // d half handled by this thread
        const int idx = lds_idx[m];
        float* outp = q_out + (size_t)b * D_ * HW + hw0 + m;
        const float* zrow = lds_z + (size_t)m * ZPAD;
        for (int j = 0; j < 64; ++j) {
            int d = dg + j;
            float e  = embed[(size_t)d * K_ + idx];
            float df = e - zrow[d];
            dsum = fmaf(df, df, dsum);
            outp[(size_t)d * HW] = e;      // consecutive threads -> consecutive m: coalesced
        }
    }
    lds_red[t] = dsum;
    __syncthreads();
    for (int s = 128; s > 0; s >>= 1) {
        if (t < s) lds_red[t] += lds_red[t + s];
        __syncthreads();
    }
    if (t == 0) partials[wg] = lds_red[0];
}

// ---------------- final deterministic diff reduce ----------------
__global__ __launch_bounds__(256) void vq_diff(const float* __restrict__ partials,
                                               float* __restrict__ out_diff)
{
    __shared__ float red[256];
    int t = threadIdx.x;
    red[t] = partials[t] + partials[t + 256];
    __syncthreads();
    for (int s = 128; s > 0; s >>= 1) {
        if (t < s) red[t] += red[t + s];
        __syncthreads();
    }
    if (t == 0) *out_diff = red[0] / (float)((size_t)N_ * D_);
}

// ---------------- launch ----------------
extern "C" void kernel_launch(void* const* d_in, const int* in_sizes, int n_in,
                              void* d_out, int out_size, void* d_ws, size_t ws_size,
                              hipStream_t stream)
{
    const float* x      = (const float*)d_in[0];
    const float* conv_w = (const float*)d_in[1];
    const float* conv_b = (const float*)d_in[2];
    const float* embed  = (const float*)d_in[3];

    float* out      = (float*)d_out;
    float* q_out    = out;                        // [B,D,H,W] = 8388608 floats
    float* diff_out = out + (size_t)N_ * D_;      // scalar
    float* ind_out  = diff_out + 1;               // [B,H,W] as float-coded ints

    char*   ws       = (char*)d_ws;
    __bf16* Wt       = (__bf16*)ws;                            // 65536 B
    __bf16* Et       = (__bf16*)(ws + 65536);                  // 262144 B
    float*  Enorm    = (float*)(ws + 65536 + 262144);          // 4096 B
    float*  partials = Enorm + K_;                             // 2048 B

    vq_prep <<<(D_ * K_ + 255) / 256, 256, 0, stream>>>(conv_w, embed, Wt, Et);
    vq_enorm<<<K_ / 256,             256, 0, stream>>>(embed, Enorm);
    vq_main <<<NWG,                  256, 0, stream>>>(x, conv_b, embed, Wt, Et,
                                                       Enorm, q_out, ind_out, partials);
    vq_diff <<<1,                    256, 0, stream>>>(partials, diff_out);
}
